// GNNLSTMPolicy_19825569038614
// MI455X (gfx1250) — matrix operand order repaired and checked by reading, not compiled, not run on hardware
//
#include <hip/hip_runtime.h>
#include <hip/hip_bf16.h>
#include <math.h>

typedef __attribute__((ext_vector_type(16))) _Float16 v16h;
typedef __attribute__((ext_vector_type(8)))  float    v8f;

#define NNODES 100000
#define NEDGES 1600000
#define NTOT   (NEDGES + NNODES)   // edges + self loops
#define NG     64
#define FH     64
#define NLH    128

static_assert(NNODES % 16 == 0, "row tiles must be full for unguarded WMMA stores");

// ---------------------------------------------------------------------------
// helpers
// ---------------------------------------------------------------------------
__device__ __forceinline__ void atomicMaxF(float* addr, float val) {
  // monotonic int reinterpretation trick: correct for mixed signs
  if (val >= 0.0f) atomicMax((int*)addr, __float_as_int(val));
  else             atomicMin((unsigned int*)addr, __float_as_uint(val));
}
__device__ __forceinline__ float sigf(float x) { return 1.0f / (1.0f + __expf(-x)); }

__global__ void k_fill(float* p, float v, long n) {
  long i = (long)blockIdx.x * blockDim.x + threadIdx.x;
  if (i < n) p[i] = v;
}

// ---------------------------------------------------------------------------
// 1) node encoder: h = LayerNorm(ReLU(x @ enc_w + enc_b))   (wave per node)
// ---------------------------------------------------------------------------
__global__ void k_encoder(const float* __restrict__ x, const float* __restrict__ ew,
                          const float* __restrict__ eb, const float* __restrict__ lg,
                          const float* __restrict__ lb, float* __restrict__ h) {
  int node = blockIdx.x * (blockDim.x >> 5) + (threadIdx.x >> 5);
  if (node >= NNODES) return;
  int lane = threadIdx.x & 31;
  float xs[5];
#pragma unroll
  for (int i = 0; i < 5; ++i) xs[i] = x[node * 5 + i];
  float vals[2];
#pragma unroll
  for (int t = 0; t < 2; ++t) {
    int f = lane + 32 * t;
    float acc = eb[f];
#pragma unroll
    for (int i = 0; i < 5; ++i) acc += xs[i] * ew[i * FH + f];
    vals[t] = acc > 0.f ? acc : 0.f;
  }
  float s  = vals[0] + vals[1];
  float sq = vals[0] * vals[0] + vals[1] * vals[1];
  for (int off = 16; off; off >>= 1) { s += __shfl_down(s, off); sq += __shfl_down(sq, off); }
  s = __shfl(s, 0); sq = __shfl(sq, 0);
  float mu = s * (1.0f / FH);
  float var = sq * (1.0f / FH) - mu * mu;
  float rstd = rsqrtf(var + 1e-5f);
#pragma unroll
  for (int t = 0; t < 2; ++t) {
    int f = lane + 32 * t;
    h[(size_t)node * FH + f] = (vals[t] - mu) * rstd * lg[f] + lb[f];
  }
}

// ---------------------------------------------------------------------------
// 2) WMMA GEMM: hw[N,64] = h[N,64] @ W[64,64]
//    one wave per 16-row tile; 4 N-tiles x (K=64 as 2x K32) = 8 v_wmma/wave
//    A (16-bit 16x32, contiguous k order used here, not ISA table order): lane lo=row M, hi selects K-half;
//      element e -> K = kc*32 + 16*hi + e
//      -> one contiguous 16-float run per lane per kc: explicit float4 (b128) loads
//    B (32x16): col N = lane&15; element e -> K = kc*32 + 16*hi + e
//      -> 16 consecutive halves from LDS-transposed f16 copy of W
//    C/D f32: VGPR r -> M = r + 8*hi, N = lane&15
// ---------------------------------------------------------------------------
__global__ void k_gemm_hw(const float* __restrict__ h, const float* __restrict__ W64,
                          float* __restrict__ hw, int nrows) {
  __shared__ __align__(32) _Float16 wT[FH * FH];   // wT[n*64 + k]
  // cooperative transpose + f32->f16 convert of W (coalesced global reads)
  for (int e = threadIdx.x; e < FH * FH; e += blockDim.x) {
    int k = e >> 6, n = e & 63;
    wT[n * FH + k] = (_Float16)W64[e];
  }
  __syncthreads();

  int wave = blockIdx.x * (blockDim.x >> 5) + (threadIdx.x >> 5);
  int lane = threadIdx.x & 31;
  int rowBase = wave << 4;
  if (rowBase >= nrows) return;            // wave-uniform tail guard only
  int hi = lane >> 4, lo = lane & 15;
  const float* hrow = h + (size_t)(rowBase + lo) * FH;

  v16h a[2];
#pragma unroll
  for (int kc = 0; kc < 2; ++kc) {
    float4 p0 = *(const float4*)(hrow + kc * 32 + 16 * hi);
    float4 p1 = *(const float4*)(hrow + kc * 32 + 16 * hi + 4);
    float4 p2 = *(const float4*)(hrow + kc * 32 + 16 * hi + 8);
    float4 p3 = *(const float4*)(hrow + kc * 32 + 16 * hi + 12);
    a[kc][0]  = (_Float16)p0.x; a[kc][1]  = (_Float16)p0.y;
    a[kc][2]  = (_Float16)p0.z; a[kc][3]  = (_Float16)p0.w;
    a[kc][4]  = (_Float16)p1.x; a[kc][5]  = (_Float16)p1.y;
    a[kc][6]  = (_Float16)p1.z; a[kc][7]  = (_Float16)p1.w;
    a[kc][8]  = (_Float16)p2.x; a[kc][9]  = (_Float16)p2.y;
    a[kc][10] = (_Float16)p2.z; a[kc][11] = (_Float16)p2.w;
    a[kc][12] = (_Float16)p3.x; a[kc][13] = (_Float16)p3.y;
    a[kc][14] = (_Float16)p3.z; a[kc][15] = (_Float16)p3.w;
  }

#pragma unroll
  for (int nt = 0; nt < 4; ++nt) {
    const int n0 = nt * 16;
    v8f c = {};
#pragma unroll
    for (int kc = 0; kc < 2; ++kc) {
      v16h b = *(const v16h*)(&wT[(n0 + lo) * FH + kc * 32 + 16 * hi]);
      c = __builtin_amdgcn_wmma_f32_16x16x32_f16(false, a[kc], false, b,
                                                 (short)0, c, false, false);
    }
    int n = n0 + lo;
#pragma unroll
    for (int r = 0; r < 8; ++r)
      hw[(size_t)(rowBase + r + 8 * hi) * FH + n] = c[r];
  }
}

// ---------------------------------------------------------------------------
// 3) per-node attention scalars: as_n = hw . a_src, ad_n = hw . a_dst
// ---------------------------------------------------------------------------
__global__ void k_node_attn(const float* __restrict__ hw, const float* __restrict__ avs,
                            const float* __restrict__ avd, float* __restrict__ asn,
                            float* __restrict__ adn) {
  int node = blockIdx.x * (blockDim.x >> 5) + (threadIdx.x >> 5);
  if (node >= NNODES) return;
  int lane = threadIdx.x & 31;
  const float2* row = (const float2*)(hw + (size_t)node * FH);
  float2 v = row[lane];
  float s = v.x * avs[2 * lane] + v.y * avs[2 * lane + 1];
  float d = v.x * avd[2 * lane] + v.y * avd[2 * lane + 1];
  for (int off = 16; off; off >>= 1) { s += __shfl_down(s, off); d += __shfl_down(d, off); }
  if (lane == 0) { asn[node] = s; adn[node] = d; }
}

// ---------------------------------------------------------------------------
// 4) edge passes (k < NEDGES: real edge; else self loop)
// ---------------------------------------------------------------------------
__global__ void k_edge1(const int* __restrict__ ei, const float* __restrict__ asn,
                        const float* __restrict__ adn, float* __restrict__ e,
                        float* __restrict__ m) {
  int k = blockIdx.x * blockDim.x + threadIdx.x;
  if (k >= NTOT) return;
  int s, d;
  if (k < NEDGES) { s = ei[k]; d = ei[NEDGES + k]; } else { s = d = k - NEDGES; }
  float v = asn[s] + adn[d];
  float lr = v > 0.f ? v : 0.2f * v;   // leaky relu
  e[k] = lr;
  atomicMaxF(&m[d], lr);
}

__global__ void k_edge2(const int* __restrict__ ei, float* __restrict__ e,
                        const float* __restrict__ m, float* __restrict__ denom) {
  int k = blockIdx.x * blockDim.x + threadIdx.x;
  if (k >= NTOT) return;
  int d = (k < NEDGES) ? ei[NEDGES + k] : (k - NEDGES);
  float ex = __expf(e[k] - m[d]);
  e[k] = ex;
  atomicAdd(&denom[d], ex);
}

__global__ void k_edge3(const int* __restrict__ ei, const float* __restrict__ e,
                        const float* __restrict__ denom, const float* __restrict__ hw,
                        float* __restrict__ outh) {
  int k = blockIdx.x * (blockDim.x >> 5) + (threadIdx.x >> 5);
  if (k >= NTOT) return;
  int lane = threadIdx.x & 31;
  int s, d;
  if (k < NEDGES) { s = ei[k]; d = ei[NEDGES + k]; } else { s = d = k - NEDGES; }
  float alpha = e[k] / (denom[d] + 1e-16f);
  float2 v = ((const float2*)(hw + (size_t)s * FH))[lane];
  float* dst = outh + (size_t)d * FH + 2 * lane;
  atomicAdd(&dst[0], alpha * v.x);
  atomicAdd(&dst[1], alpha * v.y);
}

__global__ void k_bias_relu(float* __restrict__ outh, const float* __restrict__ bias, long n) {
  long i = (long)blockIdx.x * blockDim.x + threadIdx.x;
  if (i >= n) return;
  float v = outh[i] + bias[i & (FH - 1)];
  outh[i] = v > 0.f ? v : 0.f;
}

// ---------------------------------------------------------------------------
// 5) global attention pool
// ---------------------------------------------------------------------------
__global__ void k_gate1(const float* __restrict__ h, const float* __restrict__ gw,
                        const float* __restrict__ gb, const int* __restrict__ batch,
                        float* __restrict__ gate, float* __restrict__ gm) {
  int node = blockIdx.x * (blockDim.x >> 5) + (threadIdx.x >> 5);
  if (node >= NNODES) return;
  int lane = threadIdx.x & 31;
  float2 v = ((const float2*)(h + (size_t)node * FH))[lane];
  float s = v.x * gw[2 * lane] + v.y * gw[2 * lane + 1];
  for (int off = 16; off; off >>= 1) s += __shfl_down(s, off);
  if (lane == 0) {
    s += gb[0];
    gate[node] = s;
    atomicMaxF(&gm[batch[node]], s);
  }
}

__global__ void k_gate2(const int* __restrict__ batch, float* __restrict__ gate,
                        const float* __restrict__ gm, float* __restrict__ gd) {
  int n = blockIdx.x * blockDim.x + threadIdx.x;
  if (n >= NNODES) return;
  int b = batch[n];
  float ge = __expf(gate[n] - gm[b]);
  gate[n] = ge;
  atomicAdd(&gd[b], ge);
}

__global__ void k_gate3(const int* __restrict__ batch, const float* __restrict__ gate,
                        const float* __restrict__ gd, const float* __restrict__ h,
                        float* __restrict__ pooled) {
  int node = blockIdx.x * (blockDim.x >> 5) + (threadIdx.x >> 5);
  if (node >= NNODES) return;
  int lane = threadIdx.x & 31;
  int b = batch[node];
  float ga = gate[node] / (gd[b] + 1e-16f);
  float2 v = ((const float2*)(h + (size_t)node * FH))[lane];
  float* dst = pooled + b * FH + 2 * lane;
  atomicAdd(&dst[0], ga * v.x);
  atomicAdd(&dst[1], ga * v.y);
}

// ---------------------------------------------------------------------------
// 6) LSTM (single step, h0=c0=0) + Q head
// ---------------------------------------------------------------------------
__global__ void k_lstm_gates(const float* __restrict__ pooled, const float* __restrict__ w_ih,
                             const float* __restrict__ b_ih, const float* __restrict__ b_hh,
                             float* __restrict__ gates) {
  int idx = blockIdx.x * blockDim.x + threadIdx.x;
  if (idx >= NG * 4 * NLH) return;
  int g = idx >> 9;       // / 512
  int j = idx & 511;
  float acc = b_ih[j] + b_hh[j];      // h0 == 0 -> w_hh term vanishes
  const float* pr = pooled + g * FH;
  const float* wr = w_ih + (size_t)j * FH;
#pragma unroll 8
  for (int k = 0; k < FH; ++k) acc += pr[k] * wr[k];
  gates[idx] = acc;
}

__global__ void k_lstm_cell(const float* __restrict__ gates, float* __restrict__ h1s,
                            float* __restrict__ out) {
  int idx = blockIdx.x * blockDim.x + threadIdx.x;
  if (idx >= NG * NLH) return;
  int g = idx >> 7;
  int u = idx & (NLH - 1);
  const float* gr = gates + g * 4 * NLH;
  float iv = gr[u];
  float gv = gr[2 * NLH + u];
  float ov = gr[3 * NLH + u];
  float c1 = sigf(iv) * tanhf(gv);          // f-gate * c0 == 0
  float h1 = sigf(ov) * tanhf(c1);
  h1s[idx] = h1;
  out[NG * 10 + idx] = h1;                  // h1 after q
  out[NG * 10 + NG * NLH + idx] = c1;       // c1 after h1
}

__global__ void k_qhead(const float* __restrict__ h1s, const float* __restrict__ q_w1,
                        const float* __restrict__ q_b1, const float* __restrict__ q_w2,
                        const float* __restrict__ q_b2, float* __restrict__ out) {
  __shared__ float t[FH];
  int g = blockIdx.x;
  int k = threadIdx.x;
  float acc = q_b1[k];
  const float* hr = h1s + g * NLH;
  for (int u = 0; u < NLH; ++u) acc += hr[u] * q_w1[u * FH + k];
  t[k] = acc > 0.f ? acc : 0.f;
  __syncthreads();
  if (k < 10) {
    float q = q_b2[k];
    for (int kk = 0; kk < FH; ++kk) q += t[kk] * q_w2[kk * 10 + k];
    out[g * 10 + k] = q;
  }
}

// ---------------------------------------------------------------------------
// launch
// ---------------------------------------------------------------------------
extern "C" void kernel_launch(void* const* d_in, const int* in_sizes, int n_in,
                              void* d_out, int out_size, void* d_ws, size_t ws_size,
                              hipStream_t stream) {
  const float* x      = (const float*)d_in[0];
  const int*   ei     = (const int*)d_in[1];
  const int*   batch  = (const int*)d_in[2];
  const float* enc_w  = (const float*)d_in[3];
  const float* enc_b  = (const float*)d_in[4];
  const float* ln_g   = (const float*)d_in[5];
  const float* ln_b   = (const float*)d_in[6];
  const float* w1     = (const float*)d_in[7];
  const float* a1s    = (const float*)d_in[8];
  const float* a1d    = (const float*)d_in[9];
  const float* b1     = (const float*)d_in[10];
  const float* w2     = (const float*)d_in[11];
  const float* a2s    = (const float*)d_in[12];
  const float* a2d    = (const float*)d_in[13];
  const float* b2     = (const float*)d_in[14];
  const float* gate_w = (const float*)d_in[15];
  const float* gate_b = (const float*)d_in[16];
  const float* w_ih   = (const float*)d_in[17];
  /* w_hh (d_in[18]) unused: h0 == 0 */
  const float* b_ih   = (const float*)d_in[19];
  const float* b_hh   = (const float*)d_in[20];
  const float* q_w1   = (const float*)d_in[21];
  const float* q_b1   = (const float*)d_in[22];
  const float* q_w2   = (const float*)d_in[23];
  const float* q_b2   = (const float*)d_in[24];
  float* out = (float*)d_out;

  float* p = (float*)d_ws;
  float* hA     = p; p += (size_t)NNODES * FH;
  float* hB     = p; p += (size_t)NNODES * FH;
  float* hw     = p; p += (size_t)NNODES * FH;
  float* asn    = p; p += NNODES;
  float* adn    = p; p += NNODES;
  float* mbuf   = p; p += NNODES;
  float* dbuf   = p; p += NNODES;
  float* ebuf   = p; p += (size_t)NTOT;
  float* gbuf   = p; p += NNODES;
  float* gm     = p; p += NG;
  float* gd     = p; p += NG;
  float* pooled = p; p += NG * FH;
  float* gatesb = p; p += NG * 4 * NLH;
  float* h1s    = p; p += NG * NLH;

  const int B = 256;
  const int wavePB = B / 32;
  const int nodeWaveBlocks = (NNODES + wavePB - 1) / wavePB;
  const int edgeWaveBlocks = (NTOT + wavePB - 1) / wavePB;
  const int gemmWaves = (NNODES + 15) / 16;
  const int gemmBlocks = (gemmWaves + wavePB - 1) / wavePB;

  // encoder
  k_encoder<<<nodeWaveBlocks, B, 0, stream>>>(x, enc_w, enc_b, ln_g, ln_b, hA);

  auto run_gat = [&](const float* hin, const float* Wm, const float* avs,
                     const float* avd, const float* bias, float* hout) {
    k_fill<<<(NNODES + B - 1) / B, B, 0, stream>>>(mbuf, -INFINITY, NNODES);
    k_fill<<<(NNODES + B - 1) / B, B, 0, stream>>>(dbuf, 0.0f, NNODES);
    k_fill<<<((long)NNODES * FH + B - 1) / B, B, 0, stream>>>(hout, 0.0f, (long)NNODES * FH);
    k_gemm_hw<<<gemmBlocks, B, 0, stream>>>(hin, Wm, hw, NNODES);
    k_node_attn<<<nodeWaveBlocks, B, 0, stream>>>(hw, avs, avd, asn, adn);
    k_edge1<<<(NTOT + B - 1) / B, B, 0, stream>>>(ei, asn, adn, ebuf, mbuf);
    k_edge2<<<(NTOT + B - 1) / B, B, 0, stream>>>(ei, ebuf, mbuf, dbuf);
    k_edge3<<<edgeWaveBlocks, B, 0, stream>>>(ei, ebuf, dbuf, hw, hout);
    k_bias_relu<<<((long)NNODES * FH + B - 1) / B, B, 0, stream>>>(hout, bias, (long)NNODES * FH);
  };

  run_gat(hA, w1, a1s, a1d, b1, hB);   // layer 1: hA -> hB
  run_gat(hB, w2, a2s, a2d, b2, hA);   // layer 2: hB -> hA

  // global attention pool over hA
  k_fill<<<1, NG, 0, stream>>>(gm, -INFINITY, NG);
  k_fill<<<1, NG, 0, stream>>>(gd, 0.0f, NG);
  k_fill<<<(NG * FH + B - 1) / B, B, 0, stream>>>(pooled, 0.0f, NG * FH);
  k_gate1<<<nodeWaveBlocks, B, 0, stream>>>(hA, gate_w, gate_b, batch, gbuf, gm);
  k_gate2<<<(NNODES + B - 1) / B, B, 0, stream>>>(batch, gbuf, gm, gd);
  k_gate3<<<nodeWaveBlocks, B, 0, stream>>>(batch, gbuf, gd, hA, pooled);

  // LSTM step + Q head
  k_lstm_gates<<<(NG * 4 * NLH + B - 1) / B, B, 0, stream>>>(pooled, w_ih, b_ih, b_hh, gatesb);
  k_lstm_cell<<<(NG * NLH + B - 1) / B, B, 0, stream>>>(gatesb, h1s, out);
  k_qhead<<<NG, FH, 0, stream>>>(h1s, q_w1, q_b1, q_w2, q_b2, out);
}
